// Attention_47029891891464
// MI455X (gfx1250) — matrix operand order repaired and checked by reading, not compiled, not run on hardware
//
#include <hip/hip_runtime.h>
#include <hip/hip_bf16.h>
#include <math.h>

typedef __attribute__((ext_vector_type(16))) __bf16 v16bf;
typedef __attribute__((ext_vector_type(8)))  float  v8f;

#define HDIM 1024
#define SLEN 2048
#define BATCH 32
#define MT 64            // S-rows per workgroup
#define NSUB (MT / 16)   // 4 M-subtiles of 16 rows
#define PADROW 1032      // bf16 elements per LDS A-row (pad -> spread banks for b128 reads)
#define NTHREADS 512     // 16 waves
#define NJ 4             // N-tiles per wave: 16 waves * 4 tiles * 16 cols = 1024 = HDIM

__device__ __forceinline__ unsigned short f2bf(float x) {
    unsigned int u = __float_as_uint(x);
    u += 0x7FFFu + ((u >> 16) & 1u);       // round-to-nearest-even
    return (unsigned short)(u >> 16);
}

__device__ __forceinline__ float fast_tanh(float x) {
#if __has_builtin(__builtin_amdgcn_tanhf)
    return __builtin_amdgcn_tanhf(x);      // V_TANH_F32 (CDNA5 trans op)
#else
    // tanh(x) = 1 - 2/(exp(2x)+1); robust at +-inf of exp
    float e = __expf(2.0f * x);
    return 1.0f - 2.0f * __builtin_amdgcn_rcpf(e + 1.0f);
#endif
}

// ---------------- Kernel 1: Ua [H,H] fp32 -> UaT [n][k] bf16 (tiled transpose) -------------
__global__ __launch_bounds__(256)
void ua_transconv_kernel(const float* __restrict__ Ua, unsigned short* __restrict__ UaT) {
    __shared__ float tile[32][33];
    const int tx = threadIdx.x & 31;
    const int ty = threadIdx.x >> 5;       // 0..7
    const int n0 = blockIdx.x * 32;
    const int h0 = blockIdx.y * 32;
    #pragma unroll
    for (int i = 0; i < 4; ++i) {
        int hl = ty + i * 8;
        tile[hl][tx] = Ua[(size_t)(h0 + hl) * HDIM + n0 + tx];
    }
    __syncthreads();
    #pragma unroll
    for (int i = 0; i < 4; ++i) {
        int nl = ty + i * 8;
        UaT[(size_t)(n0 + nl) * HDIM + h0 + tx] = f2bf(tile[tx][nl]);
    }
}

// ---------------- Kernel 2: dp2[b,k] = dec[b,:]@Wa[:,k] + Wa_b[k] + Ua_b[k] ----------------
__global__ __launch_bounds__(256)
void decproj_kernel(const float* __restrict__ dec, const float* __restrict__ Wa,
                    const float* __restrict__ Wa_b, const float* __restrict__ Ua_b,
                    float* __restrict__ dp2) {
    const int k = blockIdx.x * 256 + threadIdx.x;
    const int b = blockIdx.y;
    float acc = Wa_b[k] + Ua_b[k];
    const float* dr = dec + (size_t)b * HDIM;
    #pragma unroll 4
    for (int h = 0; h < HDIM; ++h)
        acc = fmaf(dr[h], Wa[(size_t)h * HDIM + k], acc);
    dp2[(size_t)b * HDIM + k] = acc;
}

// ---------------- Kernel 3: fused enc@Ua -> tanh -> dot(Va) -> masked scores ----------------
__global__ __launch_bounds__(NTHREADS)
void attn_energy_kernel(const float* __restrict__ enc,            // [B,S,H] fp32
                        const unsigned short* __restrict__ UaT,   // [H(n),H(k)] bf16
                        const float* __restrict__ dp2,            // [B,H]
                        const float* __restrict__ Va,             // [H]
                        const int*   __restrict__ mask,           // [B,S]
                        const float* __restrict__ Va_b,           // [1]
                        float* __restrict__ scores)               // [B,S]
{
    extern __shared__ char smem_raw[];
    unsigned short* At = (unsigned short*)smem_raw;               // MT x PADROW bf16
    float* sacc = (float*)(smem_raw + (size_t)MT * PADROW * 2);   // MT floats

    const int b   = blockIdx.y;
    const int s0  = blockIdx.x * MT;
    const int tid = threadIdx.x;

    if (tid < MT) sacc[tid] = 0.0f;

    // Stage A tile: MT x H fp32 -> bf16 in LDS (coalesced float4 reads)
    {
        const float4* src = (const float4*)(enc + ((size_t)b * SLEN + s0) * HDIM);
        const int n4 = MT * HDIM / 4;                 // 16384 float4
        for (int idx = tid; idx < n4; idx += NTHREADS) {
            int row = idx >> 8;                       // 256 float4 per row
            int c4  = idx & 255;
            float4 v = src[(size_t)row * 256 + c4];
            unsigned int p0 = (unsigned int)f2bf(v.x) | ((unsigned int)f2bf(v.y) << 16);
            unsigned int p1 = (unsigned int)f2bf(v.z) | ((unsigned int)f2bf(v.w) << 16);
            *(uint2*)&At[row * PADROW + c4 * 4] = make_uint2(p0, p1);
        }
    }
    __syncthreads();

    const int wave = tid >> 5;
    const int lane = tid & 31;
    const int ln   = lane & 15;
    const bool hi  = lane >= 16;

    // A-fragment bases (ISA 16-bit A layout: lanes>=16 take K+8 / K+24 halves)
    const unsigned short* a_base[NSUB];
    #pragma unroll
    for (int m = 0; m < NSUB; ++m)
        a_base[m] = At + (m * 16 + ln) * PADROW + (hi ? 8 : 0);

    // B-fragment row bases: UaT row = output column n, contiguous in k
    const unsigned short* brow[NJ];
    #pragma unroll
    for (int j = 0; j < NJ; ++j) {
        int n0 = (wave * NJ + j) * 16;
        brow[j] = UaT + (size_t)(n0 + ln) * HDIM + (hi ? 8 : 0);
    }

    v8f acc[NSUB][NJ];
    #pragma unroll
    for (int m = 0; m < NSUB; ++m)
        #pragma unroll
        for (int j = 0; j < NJ; ++j) acc[m][j] = (v8f){};

    // k outer / j inner: each A fragment loaded once from LDS, reused for NJ N-tiles;
    // each B fragment reused for NSUB M-subtiles.
    #pragma unroll 1
    for (int k0 = 0; k0 < HDIM; k0 += 32) {
        v16bf a[NSUB];
        #pragma unroll
        for (int m = 0; m < NSUB; ++m) {
            union { uint4 q[2]; v16bf v; } u;
            u.q[0] = *(const uint4*)(a_base[m] + k0);
            u.q[1] = *(const uint4*)(a_base[m] + k0 + 16);
            a[m] = u.v;
        }
        #pragma unroll
        for (int j = 0; j < NJ; ++j) {
            __builtin_prefetch(brow[j] + k0 + 256, 0, 3);
            v16bf bf;
            { union { uint4 q[2]; v16bf v; } u;
              u.q[0] = *(const uint4*)(brow[j] + k0);
              u.q[1] = *(const uint4*)(brow[j] + k0 + 16);
              bf = u.v; }
            #pragma unroll
            for (int m = 0; m < NSUB; ++m)
                acc[m][j] = __builtin_amdgcn_wmma_f32_16x16x32_bf16(false, a[m], false, bf,
                                                                    (short)0, acc[m][j],
                                                                    false, false);
        }
    }

    // Epilogue: energy = tanh(acc + dp); partial score += energy * Va[n]
    float partial[NSUB][8];
    #pragma unroll
    for (int m = 0; m < NSUB; ++m)
        #pragma unroll
        for (int r = 0; r < 8; ++r) partial[m][r] = 0.0f;

    #pragma unroll
    for (int j = 0; j < NJ; ++j) {
        const int n0 = (wave * NJ + j) * 16;
        const float dp = dp2[(size_t)b * HDIM + n0 + ln];
        const float va = Va[n0 + ln];
        #pragma unroll
        for (int m = 0; m < NSUB; ++m)
            #pragma unroll
            for (int r = 0; r < 8; ++r)
                partial[m][r] += fast_tanh(acc[m][j][r] + dp) * va;
    }

    // Reduce across the 16 N-columns held in each half-wave (C layout: halves are row groups)
    #pragma unroll
    for (int m = 0; m < NSUB; ++m)
        #pragma unroll
        for (int r = 0; r < 8; ++r) {
            float v = partial[m][r];
            v += __shfl_xor(v, 1, 32);
            v += __shfl_xor(v, 2, 32);
            v += __shfl_xor(v, 4, 32);
            v += __shfl_xor(v, 8, 32);
            partial[m][r] = v;
        }

    if (ln == 0) {           // lane 0: rows 16m + r ; lane 16: rows 16m + 8 + r
        const int roff = hi ? 8 : 0;
        #pragma unroll
        for (int m = 0; m < NSUB; ++m)
            #pragma unroll
            for (int r = 0; r < 8; ++r)
                atomicAdd(&sacc[m * 16 + roff + r], partial[m][r]);
    }
    __syncthreads();

    if (tid < MT) {
        float s = sacc[tid] + Va_b[0];
        int m = mask[(size_t)b * SLEN + s0 + tid];
        scores[(size_t)b * SLEN + s0 + tid] = (m == 0) ? -1e9f : s;
    }
}

// ---------------- Kernel 4: softmax over S per batch row ----------------
__global__ __launch_bounds__(256)
void softmax_kernel(const float* __restrict__ scores, float* __restrict__ out) {
    __shared__ float red[256];
    const int b = blockIdx.x, tid = threadIdx.x;
    const float* row = scores + (size_t)b * SLEN;
    float vals[8];
    float m = -INFINITY;
    #pragma unroll
    for (int i = 0; i < 8; ++i) { vals[i] = row[tid + i * 256]; m = fmaxf(m, vals[i]); }
    red[tid] = m; __syncthreads();
    for (int s = 128; s > 0; s >>= 1) {
        if (tid < s) red[tid] = fmaxf(red[tid], red[tid + s]);
        __syncthreads();
    }
    m = red[0]; __syncthreads();
    float sum = 0.0f;
    #pragma unroll
    for (int i = 0; i < 8; ++i) { vals[i] = __expf(vals[i] - m); sum += vals[i]; }
    red[tid] = sum; __syncthreads();
    for (int s = 128; s > 0; s >>= 1) {
        if (tid < s) red[tid] += red[tid + s];
        __syncthreads();
    }
    const float inv = 1.0f / red[0];
    #pragma unroll
    for (int i = 0; i < 8; ++i) out[(size_t)b * SLEN + tid + i * 256] = vals[i] * inv;
}

extern "C" void kernel_launch(void* const* d_in, const int* in_sizes, int n_in,
                              void* d_out, int out_size, void* d_ws, size_t ws_size,
                              hipStream_t stream) {
    (void)in_sizes; (void)n_in; (void)out_size; (void)ws_size;
    const float* dec  = (const float*)d_in[0];   // [32,1024]
    const float* enc  = (const float*)d_in[1];   // [32,2048,1024]
    const int*   mask = (const int*)  d_in[2];   // [32,2048]
    const float* Wa_w = (const float*)d_in[3];   // [1024,1024]
    const float* Wa_b = (const float*)d_in[4];   // [1024]
    const float* Ua_w = (const float*)d_in[5];   // [1024,1024]
    const float* Ua_b = (const float*)d_in[6];   // [1024]
    const float* Va_w = (const float*)d_in[7];   // [1024]
    const float* Va_b = (const float*)d_in[8];   // [1]
    float* out = (float*)d_out;

    char* ws = (char*)d_ws;
    float*          dp2    = (float*)ws;                              // 128 KB
    float*          scores = (float*)(ws + (128 << 10));              // 256 KB
    unsigned short* UaT    = (unsigned short*)(ws + (384 << 10));     // 2 MB

    ua_transconv_kernel<<<dim3(HDIM / 32, HDIM / 32), 256, 0, stream>>>(Ua_w, UaT);
    decproj_kernel<<<dim3(HDIM / 256, BATCH), 256, 0, stream>>>(dec, Wa_w, Wa_b, Ua_b, dp2);

    const size_t lds = (size_t)MT * PADROW * 2 + MT * sizeof(float);  // ~132 KB
    attn_energy_kernel<<<dim3(SLEN / MT, BATCH), NTHREADS, lds, stream>>>(
        enc, UaT, dp2, Va_w, mask, Va_b, scores);

    softmax_kernel<<<BATCH, 256, 0, stream>>>(scores, out);
}
